// VectorQuantizer_61177514164810
// MI455X (gfx1250) — matrix operand order repaired and checked by reading, not compiled, not run on hardware
//
#include <hip/hip_runtime.h>
#include <hip/hip_bf16.h>
#include <math.h>

// ---------------------------------------------------------------------------
// VQ (vector quantizer) forward for MI455X / gfx1250, wave32 + WMMA bf16.
//   M = 32768 flat rows, D = 256, K = 4096 codes.
//   score(row, k) = ||e_k||^2 - 2 * dot(z_row, e_k)   (row-constant dropped)
//   dot products via v_wmma_f32_16x16x32_bf16; B tiles staged to LDS by the
//   Tensor Data Mover (tensor_load_to_lds), double-buffered, TENSORcnt-waited.
//   B fragments double-buffered in registers (two live Frag vars, no copies)
//   so ds_load_b128 pairs overlap WMMA execution.
// ---------------------------------------------------------------------------

typedef __attribute__((ext_vector_type(16))) __bf16    v16bf;
typedef __attribute__((ext_vector_type(8)))  float     v8f;
typedef __attribute__((ext_vector_type(4)))  unsigned  v4u;
typedef __attribute__((ext_vector_type(8)))  int       v8i;
typedef __attribute__((ext_vector_type(4)))  int       v4i;

union Frag {           // 32 bytes = 8 VGPRs = one 16-bit WMMA A/B fragment
    v16bf bf;
    v4u   u[2];
};

#define VQ_M        32768
#define VQ_D        256
#define VQ_K        4096
#define VQ_NT       (VQ_K / 16)            /* 256 N-tiles */
#define VQ_QOFF     0
#define VQ_IOFF     (VQ_M * VQ_D)          /* 8388608 */
#define VQ_SOFF     (VQ_IOFF + VQ_M)       /* 8421376: loss, cb, commit, perp */
#define VQ_NWG      256                    /* main-kernel workgroups */

// LDS B-tile: 16 code rows x 512B, padded to 528B/row so the 16 lanes of a
// half hit 64 distinct banks on ds_load_b128 (528 mod 256 = 16B lane stride).
#define TILE_ROW_B  528
#define TILE_BYTES  (16 * TILE_ROW_B)      /* 8448 B per buffer */

#if __has_builtin(__builtin_amdgcn_tensor_load_to_lds) && \
    __has_builtin(__builtin_amdgcn_s_wait_tensorcnt)
#define VQ_HAVE_TDM 1
#else
#define VQ_HAVE_TDM 0
#endif

// ------------------------- kernel 1: z fp32 -> bf16 -------------------------
__global__ __launch_bounds__(256)
void vq_convert_z(const float* __restrict__ z, unsigned short* __restrict__ z_bf)
{
    // 4096 blocks * 256 threads * 8 elements = 8,388,608
    const int base = (blockIdx.x * 256 + threadIdx.x) * 8;
    const float4 a = *(const float4*)(z + base);
    const float4 b = *(const float4*)(z + base + 4);
    union { __bf16 h[8]; v4u u; } pk;
    pk.h[0] = (__bf16)a.x; pk.h[1] = (__bf16)a.y;
    pk.h[2] = (__bf16)a.z; pk.h[3] = (__bf16)a.w;
    pk.h[4] = (__bf16)b.x; pk.h[5] = (__bf16)b.y;
    pk.h[6] = (__bf16)b.z; pk.h[7] = (__bf16)b.w;
    *(v4u*)(z_bf + base) = pk.u;
}

// ---------- kernel 2: embed fp32 -> bf16, ||e||^2 (fp32), zero counts -------
__global__ __launch_bounds__(256)
void vq_embed_prep(const float* __restrict__ embed,
                   unsigned short* __restrict__ e_bf,
                   float* __restrict__ enorm,
                   unsigned* __restrict__ counts)
{
    __shared__ float s_wave[8];
    const int row = blockIdx.x;           // 4096 blocks, one codebook row each
    const int t   = threadIdx.x;          // 256 threads = 256 dims
    const float v = embed[row * VQ_D + t];
    ((__bf16*)e_bf)[row * VQ_D + t] = (__bf16)v;
    float sq = v * v;
    #pragma unroll
    for (int off = 1; off < 32; off <<= 1) sq += __shfl_xor(sq, off);
    if ((t & 31) == 0) s_wave[t >> 5] = sq;
    __syncthreads();
    if (t == 0) {
        float s = 0.f;
        #pragma unroll
        for (int i = 0; i < 8; ++i) s += s_wave[i];
        enorm[row]  = s;
        counts[row] = 0u;                 // exactly one block per count slot
    }
}

#if VQ_HAVE_TDM
// Issue one TDM load: 16x256 bf16 tile of the [4096,256] codebook tensor into
// LDS at lds_off, with 16B padding inserted every 512B (row stride 528B).
// D# layout per CDNA5 ISA.  This toolchain's builtin is
// the 6-arg form: (v4u g0, v8i g1, v4i g2, v4i g3, v8i g4, i32 cpol).
__device__ __forceinline__ void vq_tdm_load_tile(unsigned lds_off,
                                                 unsigned long long gaddr)
{
    const v4u g0 = {
        1u,                                            // count=1, user D#
        lds_off,                                       // lds_addr
        (unsigned)(gaddr & 0xffffffffu),               // global_addr[31:0]
        ((unsigned)(gaddr >> 32) & 0x01ffffffu)        // global_addr[56:32]
            | 0x80000000u                              // type=2 ("image")
    };
    const v8i g1 = {
        (int)((1u << 16)          // data_size = 1 -> 2 bytes
            | (1u << 20)          // pad_enable
            | (6u << 22)          // pad_interval = 6 -> every 128 dwords (512B)
            | (3u << 25)),        // pad_amount   = 3 -> 4 dwords (16B)
        (int)(256u << 16),        // tensor_dim0 = 256     (bits 79:48)
        (int)(4096u << 16),       // tensor_dim1 = 4096    (bits 111:80)
        (int)(256u << 16),        // tile_dim0 = 256       (bits 127:112)
        16,                       // tile_dim1 = 16
        256,                      // tensor_dim0_stride = 256
        0, 0
    };
    const v4i gz4 = {0, 0, 0, 0};
    const v8i gz8 = {0, 0, 0, 0, 0, 0, 0, 0};
    __builtin_amdgcn_tensor_load_to_lds(g0, g1, gz4, gz4, gz8, 0);
}
#endif

// ------------------- kernel 3: WMMA GEMM + argmin + epilogue ----------------
__global__ __launch_bounds__(256)
void vq_main(const float* __restrict__ z,
             const float* __restrict__ embed,
             const unsigned short* __restrict__ z_bf_,
             const unsigned short* __restrict__ e_bf_,
             const float* __restrict__ enorm,
             unsigned* __restrict__ counts,
             float* __restrict__ partials,
             float* __restrict__ out)
{
    __shared__ __align__(16) char s_bt[2][TILE_BYTES];   // double-buffered B
    __shared__ int   s_idx[128];          // per-WG winning index per row
    __shared__ float s_wsum[8];

    const __bf16* z_bf = (const __bf16*)z_bf_;

    const int tid = threadIdx.x;
    const int w   = tid >> 5;             // wave id 0..7
    const int l   = tid & 31;             // lane
    const int lh  = l & 15;               // lane-in-half
    const int kh  = l >> 4;               // half id (K-group select)
    const int m0  = blockIdx.x * 128 + w * 16;   // this wave's 16 rows
    const int mrow = m0 + lh;

    // -- A fragments: 16x32 bf16 per k-step, 8 steps cover D=256. Layout per
    //    ISA 7.12.2: lanes 0-15 -> M=lane, v0-3 K=8*kh+0..7, v4-7 K=+16.
    Frag A[8];
    const __bf16* zrow = z_bf + (size_t)mrow * VQ_D + 8 * kh;
    #pragma unroll
    for (int s = 0; s < 8; ++s) {
        A[s].u[0] = *(const v4u*)(zrow + 32 * s);
        A[s].u[1] = *(const v4u*)(zrow + 32 * s + 16);
    }

    float best[8];
    int   bidx[8];
    #pragma unroll
    for (int v = 0; v < 8; ++v) { best[v] = 3.4e38f; bidx[v] = 0; }

#if VQ_HAVE_TDM
    const unsigned long long ebase = (unsigned long long)(size_t)e_bf_;
    const unsigned lds0 = (unsigned)(size_t)(void*)&s_bt[0][0];
    const unsigned lds1 = (unsigned)(size_t)(void*)&s_bt[1][0];
    if (w == 0)                                   // prologue: tile 0 -> buf 0
        vq_tdm_load_tile(lds0, ebase);
#endif

    for (int t = 0; t < VQ_NT; ++t) {             // 256 N-tiles of 16 codes
        const int pb = t & 1;
#if VQ_HAVE_TDM
        if (w == 0)
            __builtin_amdgcn_s_wait_tensorcnt(0); // tile t landed in LDS
        __syncthreads();                          // publish to all 8 waves
        if (w == 0 && t + 1 < VQ_NT)              // prefetch tile t+1
            vq_tdm_load_tile((t & 1) ? lds0 : lds1,
                             ebase + (unsigned long long)(t + 1) * (16 * 512));
#else
        __syncthreads();
        {   // cooperative copy: 256 threads x 32B = 8KB tile, padded rows
            const int rr = tid >> 4, cc = (tid & 15) * 32;
            const char* g = (const char*)e_bf_ + (size_t)(t * 16 + rr) * 512 + cc;
            char* d = s_bt[pb] + rr * TILE_ROW_B + cc;
            *(v4u*)(d)      = *(const v4u*)(g);
            *(v4u*)(d + 16) = *(const v4u*)(g + 16);
        }
        __syncthreads();
#endif
        // -- B fragments from LDS (32x16 bf16): v0-3 K=8*kh+0..7,
        //    v4-7 K=16+8*kh+0..7 (matches A); lane-in-half = N column.
        //    Two live fragment buffers, no copies: loads for one buffer are
        //    in flight while the WMMA consuming the other executes.
        const char* bb = s_bt[pb] + lh * TILE_ROW_B + 16 * kh;
        v8f acc = {0.f, 0.f, 0.f, 0.f, 0.f, 0.f, 0.f, 0.f};
        Frag B0, B1;
        B0.u[0] = *(const v4u*)(bb);
        B0.u[1] = *(const v4u*)(bb + 32);
        #pragma unroll
        for (int s = 0; s < 8; s += 2) {
            B1.u[0] = *(const v4u*)(bb + 64 * (s + 1));
            B1.u[1] = *(const v4u*)(bb + 64 * (s + 1) + 32);
            acc = __builtin_amdgcn_wmma_f32_16x16x32_bf16(
                      false, A[s].bf, false, B0.bf, (short)0, acc, false, false);
            if (s + 2 < 8) {
                B0.u[0] = *(const v4u*)(bb + 64 * (s + 2));
                B0.u[1] = *(const v4u*)(bb + 64 * (s + 2) + 32);
            }
            acc = __builtin_amdgcn_wmma_f32_16x16x32_bf16(
                      false, A[s + 1].bf, false, B1.bf, (short)0, acc, false, false);
        }
        // C layout: VGPR v -> row v + 8*kh, lane-in-half -> column.
        const int   col = t * 16 + lh;
        const float en  = enorm[col];
        #pragma unroll
        for (int v = 0; v < 8; ++v) {
            const float sc = en - 2.0f * acc[v];
            if (sc < best[v]) { best[v] = sc; bidx[v] = col; }
        }
    }

    // -- argmin across the 16 lanes of each half (ties -> lower index).
    #pragma unroll
    for (int v = 0; v < 8; ++v) {
        float b = best[v];
        int   i = bidx[v];
        #pragma unroll
        for (int off = 1; off < 16; off <<= 1) {
            const float ob = __shfl_xor(b, off);
            const int   oi = __shfl_xor(i, off);
            if (ob < b || (ob == b && oi < i)) { b = ob; i = oi; }
        }
        if (lh == v) s_idx[w * 16 + v + 8 * kh] = i;   // one lane per row
    }
    __syncthreads();

    // -- epilogue: gather fp32 code rows, commitment partial, histogram.
    float csum = 0.f;
    #pragma unroll 1
    for (int r = 0; r < 16; ++r) {
        const int row = m0 + r;
        const int idx = s_idx[w * 16 + r];
        const float4* e4 = (const float4*)(embed + (size_t)idx * VQ_D);
        const float4* z4 = (const float4*)(z     + (size_t)row * VQ_D);
        float4*       o4 = (float4*)(out + VQ_QOFF + (size_t)row * VQ_D);
        #pragma unroll
        for (int q = 0; q < 2; ++q) {
            const float4 e  = e4[l * 2 + q];
            const float4 zz = z4[l * 2 + q];
            o4[l * 2 + q] = e;                          // quantized_st == quantized
            const float dx = e.x - zz.x, dy = e.y - zz.y;
            const float dz = e.z - zz.z, dw = e.w - zz.w;
            csum += dx * dx + dy * dy + dz * dz + dw * dw;
        }
        if (l == 0) {
            out[VQ_IOFF + row] = (float)idx;            // indices (row-major)
            atomicAdd(&counts[idx], 1u);                // deterministic (int)
        }
    }
    #pragma unroll
    for (int off = 1; off < 32; off <<= 1) csum += __shfl_xor(csum, off);
    if (l == 0) s_wsum[w] = csum;
    __syncthreads();
    if (tid == 0) {
        float s = 0.f;
        #pragma unroll
        for (int i = 0; i < 8; ++i) s += s_wsum[i];
        partials[blockIdx.x] = s;                       // fixed-order finalize
    }
}

// ----------------- kernel 4: losses + perplexity (1 block) ------------------
__global__ __launch_bounds__(256)
void vq_finalize(const unsigned* __restrict__ counts,
                 const float* __restrict__ partials,
                 float* __restrict__ out)
{
    __shared__ float red[256];
    const int t = threadIdx.x;

    red[t] = partials[t];                 // 256 per-WG commitment partials
    __syncthreads();
    for (int s = 128; s > 0; s >>= 1) {
        if (t < s) red[t] += red[t + s];
        __syncthreads();
    }
    const float csum = red[0];
    __syncthreads();

    float ent = 0.f;
    #pragma unroll
    for (int i = 0; i < 16; ++i) {
        const float p = (float)counts[t * 16 + i] * (1.0f / (float)VQ_M);
        ent += p * logf(p + 1e-10f);
    }
    red[t] = ent;
    __syncthreads();
    for (int s = 128; s > 0; s >>= 1) {
        if (t < s) red[t] += red[t + s];
        __syncthreads();
    }
    if (t == 0) {
        const float commitment = csum / (float)(VQ_M * VQ_D);
        out[VQ_SOFF + 0] = 0.25f * commitment;   // loss (codebook term is 0)
        out[VQ_SOFF + 1] = 0.0f;                 // codebook_loss
        out[VQ_SOFF + 2] = commitment;           // commitment_loss
        out[VQ_SOFF + 3] = expf(-red[0]);        // perplexity
    }
}

// ---------------------------------------------------------------------------
extern "C" void kernel_launch(void* const* d_in, const int* in_sizes, int n_in,
                              void* d_out, int out_size, void* d_ws, size_t ws_size,
                              hipStream_t stream)
{
    (void)in_sizes; (void)n_in; (void)out_size; (void)ws_size;
    const float* z     = (const float*)d_in[0];   // [8192, 1024] fp32
    const float* embed = (const float*)d_in[1];   // [4096, 256]  fp32
    float*       out   = (float*)d_out;

    char* ws = (char*)d_ws;                       // ~18.9 MB used
    unsigned short* z_bf     = (unsigned short*)(ws);              // 16,777,216 B
    unsigned short* e_bf     = (unsigned short*)(ws + 16777216);   //  2,097,152 B
    float*          enorm    = (float*)(ws + 18874368);            //     16,384 B
    unsigned*       counts   = (unsigned*)(ws + 18890752);         //     16,384 B
    float*          partials = (float*)(ws + 18907136);            //      1,024 B

    vq_convert_z <<<4096,   256, 0, stream>>>(z, z_bf);
    vq_embed_prep<<<VQ_K,   256, 0, stream>>>(embed, e_bf, enorm, counts);
    vq_main      <<<VQ_NWG, 256, 0, stream>>>(z, embed, z_bf, e_bf, enorm,
                                              counts, partials, out);
    vq_finalize  <<<1,      256, 0, stream>>>(counts, partials, out);
}
